// GCN_10917806866525
// MI455X (gfx1250) — matrix operand order repaired and checked by reading, not compiled, not run on hardware
//
#include <hip/hip_runtime.h>
#include <math.h>

#define NNODES 100000
#define NEDGES 1600000
#define NEP    1700000   // edges + self loops
#define NGRAPH 64
#define FIN    128

typedef _Float16 v16h __attribute__((ext_vector_type(16)));
typedef _Float16 v8h  __attribute__((ext_vector_type(8)));
typedef float    v8f  __attribute__((ext_vector_type(8)));

// ---------------- atomics ----------------
__device__ __forceinline__ void atomic_add_f32(float* addr, float v) {
    __hip_atomic_fetch_add(addr, v, __ATOMIC_RELAXED, __HIP_MEMORY_SCOPE_AGENT);
}

__device__ __forceinline__ void atomic_max_f32(float* addr, float val) {
    // works with init = -FLT_MAX; positive floats ordered as ints,
    // negative floats reverse-ordered as uints
    if (val >= 0.0f) atomicMax((int*)addr, __float_as_int(val));
    else             atomicMin((unsigned int*)addr, __float_as_uint(val));
}

// ---------------- precision prep ----------------
__global__ void f32_to_f16(const float* __restrict__ X, _Float16* __restrict__ Y, int n) {
    int i = blockIdx.x * blockDim.x + threadIdx.x;
    int stride = gridDim.x * blockDim.x;
    for (; i < n; i += stride) Y[i] = (_Float16)X[i];
}

// Re-layout W[K x H] (row major f32) into fragment order fp16:
//   Wh[(col * (K/16) + kblock)*16 + r] = W[(kblock*16 + r)*H + col]
// so a B fragment's 16 K-values for one column are contiguous (2x b128).
__global__ void w_frag_f16(const float* __restrict__ W, _Float16* __restrict__ Wh,
                           int K, int H) {
    int i = blockIdx.x * blockDim.x + threadIdx.x;
    if (i < K * H) {
        int k = i / H, c = i % H;
        int kb = k >> 4, r = k & 15;
        Wh[((size_t)c * (K >> 4) + kb) * 16 + r] = (_Float16)W[i];
    }
}

// ---------------- dense transforms via WMMA ----------------
// One wave computes one 16x16 tile of BOTH  XL = X@Wl+bl  and  XR = X@Wr+br.
// A fragment (16-bit A 16x32, ISA 7.12.2): lane l holds row M=l&15,
//   K = (l>=16)*16 + i, i=0..15 (contiguous, same k order as B).
// B fragment: lane l holds col N=l&15, K = (l>=16)*16 + i, i=0..15
//   (served contiguously from the fragment-ordered fp16 weight buffer).
// C/D: VGPR d -> row d (lanes 0-15) / d+8 (lanes 16-31), col = lane&15.
template <int K, int H>
__global__ void __launch_bounds__(128)
gatv2_dense_wmma(const _Float16* __restrict__ Xh,
                 const _Float16* __restrict__ Wlh, const float* __restrict__ bl,
                 const _Float16* __restrict__ Wrh, const float* __restrict__ br,
                 float* __restrict__ XL, float* __restrict__ XR)
{
    constexpr int rowTiles = NNODES / 16;
    constexpr int colTiles = H / 16;
    constexpr int totalTiles = rowTiles * colTiles;
    constexpr int KB = K >> 4;

    const int lane = threadIdx.x & 31;
    const int tile = blockIdx.x * 4 + (threadIdx.x >> 5);
    if (tile >= totalTiles) return;   // uniform per wave -> EXEC all-ones for WMMA

    const int rowBase = (tile / colTiles) * 16;
    const int colBase = (tile % colTiles) * 16;

    const int m   = lane & 15;
    const int hiA = (lane >> 4) << 4;   // 0 or 16 (half elements)
    const int hiK = (lane >> 4);        // 0 or 1  (extra 16-K block for B)
    const int row = rowBase + m;
    const int col = colBase + m;

    v8f accL = {};
    v8f accR = {};

    #pragma unroll
    for (int k0 = 0; k0 < K; k0 += 32) {
        // ---- A fragment: two contiguous 8-half runs -> 2x global_load_b128 ----
        const v8h* pa = reinterpret_cast<const v8h*>(Xh + (size_t)row * K + k0 + hiA);
        v8h alo = pa[0];      // K = k0+hiA   .. +7
        v8h ahi = pa[1];      // K = k0+8+hiA .. +7
        v16h a;
        #pragma unroll
        for (int i = 0; i < 8; ++i) { a[i] = alo[i]; a[8 + i] = ahi[i]; }

        // ---- B fragments: contiguous 16-half run per lane -> 2x b128 each ----
        const int kb = (k0 >> 4) + hiK;
        const v8h* pbl = reinterpret_cast<const v8h*>(Wlh + ((size_t)col * KB + kb) * 16);
        const v8h* pbr = reinterpret_cast<const v8h*>(Wrh + ((size_t)col * KB + kb) * 16);
        v8h bl0 = pbl[0], bl1 = pbl[1];
        v8h br0 = pbr[0], br1 = pbr[1];
        v16h bL, bR;
        #pragma unroll
        for (int i = 0; i < 8; ++i) {
            bL[i] = bl0[i]; bL[8 + i] = bl1[i];
            bR[i] = br0[i]; bR[8 + i] = br1[i];
        }

        accL = __builtin_amdgcn_wmma_f32_16x16x32_f16(false, a, false, bL,
                                                      (short)0, accL, false, false);
        accR = __builtin_amdgcn_wmma_f32_16x16x32_f16(false, a, false, bR,
                                                      (short)0, accR, false, false);
    }

    const int n  = lane & 15;
    const int m0 = (lane >> 4) << 3;
    const float addL = bl[colBase + n];
    const float addR = br[colBase + n];
    #pragma unroll
    for (int d = 0; d < 8; ++d) {
        size_t idx = (size_t)(rowBase + m0 + d) * H + colBase + n;
        XL[idx] = accL[d] + addL;
        XR[idx] = accR[d] + addR;
    }
}

// ---------------- edge kernels ----------------
// e = att . leaky_relu(xl[src] + xr[dst]); running segment max into emax[dst]
__global__ void edge_logits(const float* __restrict__ XL, const float* __restrict__ XR,
                            const int* __restrict__ ei, const float* __restrict__ att,
                            float* __restrict__ ew, float* __restrict__ emax, int H)
{
    const int lane = threadIdx.x & 31;
    const int wpb  = blockDim.x >> 5;
    int wid = blockIdx.x * wpb + (threadIdx.x >> 5);
    int stride = gridDim.x * wpb;
    for (int e = wid; e < NEP; e += stride) {
        int s, d;
        if (e < NEDGES) { s = ei[e]; d = ei[NEDGES + e]; }
        else            { s = d = e - NEDGES; }
        float part = 0.f;
        for (int h = lane; h < H; h += 32) {
            float v = XL[(size_t)s * H + h] + XR[(size_t)d * H + h];
            v = (v > 0.f) ? v : 0.2f * v;
            part += v * att[h];
        }
        #pragma unroll
        for (int off = 16; off > 0; off >>= 1) part += __shfl_xor(part, off, 32);
        if (lane == 0) {
            ew[e] = part;
            atomic_max_f32(&emax[d], part);
        }
    }
}

// w = exp(e - emax[dst]); denom[dst] += w
__global__ void edge_exp(const int* __restrict__ ei, float* __restrict__ ew,
                         const float* __restrict__ emax, float* __restrict__ denom)
{
    int i = blockIdx.x * blockDim.x + threadIdx.x;
    int stride = gridDim.x * blockDim.x;
    for (int e = i; e < NEP; e += stride) {
        int d = (e < NEDGES) ? ei[NEDGES + e] : (e - NEDGES);
        float w = __expf(ew[e] - emax[d]);
        ew[e] = w;
        atomic_add_f32(&denom[d], w);
    }
}

// out[dst] += (w/denom[dst]) * xl[src]
__global__ void edge_aggregate(const float* __restrict__ XL, const int* __restrict__ ei,
                               const float* __restrict__ ew, const float* __restrict__ denom,
                               float* __restrict__ out, int H)
{
    const int lane = threadIdx.x & 31;
    const int wpb  = blockDim.x >> 5;
    int wid = blockIdx.x * wpb + (threadIdx.x >> 5);
    int stride = gridDim.x * wpb;
    for (int e = wid; e < NEP; e += stride) {
        int s, d;
        if (e < NEDGES) { s = ei[e]; d = ei[NEDGES + e]; }
        else            { s = d = e - NEDGES; }
        float alpha = ew[e] / (denom[d] + 1e-16f);
        for (int h = lane; h < H; h += 32)
            atomic_add_f32(&out[(size_t)d * H + h], alpha * XL[(size_t)s * H + h]);
    }
}

// ---------------- batchnorm ----------------
__global__ void bn_stats(const float* __restrict__ X, float* __restrict__ mu,
                         float* __restrict__ rstd, int H)
{
    __shared__ float s_sum[256], s_sq[256];
    const int h = blockIdx.x;
    const int tid = threadIdx.x;
    float sum = 0.f, sq = 0.f;
    for (int n = tid; n < NNODES; n += 256) {
        float v = X[(size_t)n * H + h];
        sum += v; sq += v * v;
    }
    s_sum[tid] = sum; s_sq[tid] = sq;
    __syncthreads();
    for (int o = 128; o > 0; o >>= 1) {
        if (tid < o) { s_sum[tid] += s_sum[tid + o]; s_sq[tid] += s_sq[tid + o]; }
        __syncthreads();
    }
    if (tid == 0) {
        float m = s_sum[0] * (1.0f / NNODES);
        float var = s_sq[0] * (1.0f / NNODES) - m * m;
        mu[h] = m;
        rstd[h] = rsqrtf(var + 1e-5f);
    }
}

// BN + ReLU; writes fp32 (for pooling on last layer) and fp16 (next GEMM input)
__global__ void bn_relu(const float* __restrict__ X, const float* __restrict__ mu,
                        const float* __restrict__ rstd, const float* __restrict__ gamma,
                        const float* __restrict__ beta, float* __restrict__ Y,
                        _Float16* __restrict__ Yh, int H)
{
    int i = blockIdx.x * blockDim.x + threadIdx.x;
    const int total = NNODES * H;
    const int stride = gridDim.x * blockDim.x;
    for (; i < total; i += stride) {
        int h = i % H;
        float v = (X[i] - mu[h]) * rstd[h] * gamma[h] + beta[h];
        v = fmaxf(v, 0.f);
        Y[i] = v;
        Yh[i] = (_Float16)v;
    }
}

// ---------------- pooling + linear ----------------
__global__ void pool_sum(const float* __restrict__ H3, const int* __restrict__ batch,
                         float* __restrict__ pooled, float* __restrict__ cnt)
{
    int i = blockIdx.x * blockDim.x + threadIdx.x;
    const int stride = gridDim.x * blockDim.x;
    for (; i < NNODES * 16; i += stride) {
        int n = i >> 4, h = i & 15;
        int g = batch[n];
        atomic_add_f32(&pooled[g * 16 + h], H3[i]);
        if (h == 0) atomic_add_f32(&cnt[g], 1.0f);
    }
}

__global__ void final_linear(const float* __restrict__ pooled, const float* __restrict__ cnt,
                             const float* __restrict__ linW, const float* __restrict__ linb,
                             float* __restrict__ out)
{
    int g = threadIdx.x;
    if (g < NGRAPH) {
        float c = fmaxf(cnt[g], 1.0f);
        float s = 0.f;
        #pragma unroll
        for (int h = 0; h < 16; ++h) s += pooled[g * 16 + h] * linW[h];
        out[g] = s / c + linb[0];
    }
}

// ---------------- init helpers ----------------
__global__ void fill_f32(float* p, float v, int n) {
    int i = blockIdx.x * blockDim.x + threadIdx.x;
    int stride = gridDim.x * blockDim.x;
    for (; i < n; i += stride) p[i] = v;
}
__global__ void init_bias_rows(float* p, const float* __restrict__ bias, int H) {
    int i = blockIdx.x * blockDim.x + threadIdx.x;
    const int total = NNODES * H;
    int stride = gridDim.x * blockDim.x;
    for (; i < total; i += stride) p[i] = bias[i % H];
}

// ---------------- launcher ----------------
extern "C" void kernel_launch(void* const* d_in, const int* in_sizes, int n_in,
                              void* d_out, int out_size, void* d_ws, size_t ws_size,
                              hipStream_t stream) {
    (void)in_sizes; (void)n_in; (void)out_size; (void)ws_size;
    const float* x     = (const float*)d_in[0];
    const int*   ei    = (const int*)d_in[1];
    const int*   batch = (const int*)d_in[2];
    const float* linW  = (const float*)d_in[27];
    const float* linb  = (const float*)d_in[28];

    char* ws = (char*)d_ws;
    size_t off = 0;
    auto carveB = [&](size_t bytes) {
        void* p = ws + off;
        off = (off + bytes + 255) & ~(size_t)255;
        return p;
    };
    float*    XL     = (float*)carveB((size_t)NNODES * 64 * 4);
    float*    XR     = (float*)carveB((size_t)NNODES * 64 * 4);
    float*    OUT    = (float*)carveB((size_t)NNODES * 64 * 4);
    float*    HB     = (float*)carveB((size_t)NNODES * 64 * 4);
    _Float16* XH     = (_Float16*)carveB((size_t)NNODES * FIN * 2);  // fp16 input x
    _Float16* HBh    = (_Float16*)carveB((size_t)NNODES * 64 * 2);   // fp16 activations
    _Float16* WLh    = (_Float16*)carveB((size_t)FIN * 64 * 2);
    _Float16* WRh    = (_Float16*)carveB((size_t)FIN * 64 * 2);
    float*    EW     = (float*)carveB((size_t)NEP * 4);
    float*    EMAX   = (float*)carveB((size_t)NNODES * 4);
    float*    DENOM  = (float*)carveB((size_t)NNODES * 4);
    float*    MU     = (float*)carveB(64 * 4);
    float*    RSTD   = (float*)carveB(64 * 4);
    float*    POOLED = (float*)carveB(NGRAPH * 16 * 4);
    float*    CNT    = (float*)carveB(NGRAPH * 4);

    // one-time fp16 copy of input features
    f32_to_f16<<<2048, 256, 0, stream>>>(x, XH, NNODES * FIN);

    const int Ks[3] = {FIN, 64, 32};
    const int Hs[3] = {64, 32, 16};
    const _Float16* Xin_h = XH;

    for (int l = 0; l < 3; ++l) {
        const int K = Ks[l], H = Hs[l];
        const float* Wl    = (const float*)d_in[3 + l * 8 + 0];
        const float* bl    = (const float*)d_in[3 + l * 8 + 1];
        const float* Wr    = (const float*)d_in[3 + l * 8 + 2];
        const float* br    = (const float*)d_in[3 + l * 8 + 3];
        const float* att   = (const float*)d_in[3 + l * 8 + 4];
        const float* bias  = (const float*)d_in[3 + l * 8 + 5];
        const float* gamma = (const float*)d_in[3 + l * 8 + 6];
        const float* beta  = (const float*)d_in[3 + l * 8 + 7];

        // weight prep: fragment-ordered fp16 (tiny)
        const int wElems = K * H;
        w_frag_f16<<<(wElems + 255) / 256, 256, 0, stream>>>(Wl, WLh, K, H);
        w_frag_f16<<<(wElems + 255) / 256, 256, 0, stream>>>(Wr, WRh, K, H);

        // dense transforms (WMMA), one wave per 16x16 tile, 4 waves/block
        const int totalTiles = (NNODES / 16) * (H / 16);
        const int gemmBlocks = (totalTiles + 3) / 4;
        if (l == 0)      gatv2_dense_wmma<FIN, 64><<<gemmBlocks, 128, 0, stream>>>(Xin_h, WLh, bl, WRh, br, XL, XR);
        else if (l == 1) gatv2_dense_wmma<64,  32><<<gemmBlocks, 128, 0, stream>>>(Xin_h, WLh, bl, WRh, br, XL, XR);
        else             gatv2_dense_wmma<32,  16><<<gemmBlocks, 128, 0, stream>>>(Xin_h, WLh, bl, WRh, br, XL, XR);

        // per-call init (graph replay safe)
        fill_f32<<<512, 256, 0, stream>>>(EMAX, -3.402823466e+38f, NNODES);
        fill_f32<<<512, 256, 0, stream>>>(DENOM, 0.f, NNODES);
        init_bias_rows<<<2048, 256, 0, stream>>>(OUT, bias, H);

        // attention softmax + aggregate (all L2-resident scatter/gather)
        edge_logits<<<4096, 256, 0, stream>>>(XL, XR, ei, att, EW, EMAX, H);
        edge_exp<<<4096, 256, 0, stream>>>(ei, EW, EMAX, DENOM);
        edge_aggregate<<<4096, 256, 0, stream>>>(XL, ei, EW, DENOM, OUT, H);

        // batchnorm + relu (writes fp32 for pooling + fp16 for next GEMM)
        bn_stats<<<H, 256, 0, stream>>>(OUT, MU, RSTD, H);
        bn_relu<<<2048, 256, 0, stream>>>(OUT, MU, RSTD, gamma, beta, HB, HBh, H);
        Xin_h = HBh;
    }

    // global mean pool + final linear
    fill_f32<<<4, 256, 0, stream>>>(POOLED, 0.f, NGRAPH * 16);
    fill_f32<<<1, 64, 0, stream>>>(CNT, 0.f, NGRAPH);
    pool_sum<<<2048, 256, 0, stream>>>(HB, batch, POOLED, CNT);
    final_linear<<<1, 64, 0, stream>>>(POOLED, CNT, linW, linb, (float*)d_out);
}
